// W8A8BFP32OFP32_GeLu_Q_36447092474288
// MI455X (gfx1250) — 8-bit matrix operand order repaired and checked by reading, compiled, not run on hardware
//
#include <hip/hip_runtime.h>
#include <stdint.h>


typedef int v8i __attribute__((ext_vector_type(8)));
typedef int v4i __attribute__((ext_vector_type(4)));
typedef int v2i __attribute__((ext_vector_type(2)));

#define TOKENS 8192
#define KDIM   4096
#define ODIM   4096

#define BM 128
#define BN 128
#define BK 64
#define NKSTEPS (KDIM / BK)
#define LDSTRIDE 80

__device__ __forceinline__ uint32_t pack4(v4i v) {
  return (uint32_t)(v.x & 0xff)        |
         ((uint32_t)(v.y & 0xff) << 8) |
         ((uint32_t)(v.z & 0xff) << 16)|
         ((uint32_t)(v.w & 0xff) << 24);
}

__device__ __forceinline__ void load_raw(const int* __restrict__ src, int tid,
                                         v4i raw[8]) {
  const int r    = tid >> 1;
  const int half = tid & 1;
  const v4i* s = reinterpret_cast<const v4i*>(src + (size_t)r * KDIM + half * 32);
#pragma unroll
  for (int i = 0; i < 8; ++i) raw[i] = s[i];
}

__device__ __forceinline__ void pack_store(const v4i raw[8],
                                           unsigned char* __restrict__ lds,
                                           int tid) {
  const int r    = tid >> 1;
  const int half = tid & 1;
  uint32_t p[8];
#pragma unroll
  for (int i = 0; i < 8; ++i) p[i] = pack4(raw[i]);
  v4i* d = reinterpret_cast<v4i*>(lds + r * LDSTRIDE + half * 32);
  v4i d0 = { (int)p[0], (int)p[1], (int)p[2], (int)p[3] };
  v4i d1 = { (int)p[4], (int)p[5], (int)p[6], (int)p[7] };
  d[0] = d0; d[1] = d1;
}

extern "C" __global__ __launch_bounds__(256)
void w8a8_gelu_q_wmma(const int*   __restrict__ x,
                      const int*   __restrict__ w,
                      const float* __restrict__ bias,
                      const float* __restrict__ pa,
                      const float* __restrict__ pb,
                      float*       __restrict__ out) {
  __shared__ __align__(16) unsigned char ldsA[2][BM * LDSTRIDE];
  __shared__ __align__(16) unsigned char ldsB[2][BN * LDSTRIDE];

  const int tid  = threadIdx.x;
  const int lane = tid & 31;
  const int wave = tid >> 5;
  const int wm   = wave >> 2;
  const int wn   = wave & 3;
  const int hi   = lane >> 4;
  const int lm   = lane & 15;

  const int tileN = blockIdx.x * BN;
  const int tileM = blockIdx.y * BM;

  v8i acc[4][2];
  const v8i zero = {};
#pragma unroll
  for (int i = 0; i < 4; ++i)
#pragma unroll
    for (int j = 0; j < 2; ++j) acc[i][j] = zero;

  const int* xA = x + (size_t)tileM * KDIM;
  const int* wB = w + (size_t)tileN * KDIM;

  v4i rawA[8], rawB[8];
  load_raw(xA, tid, rawA);
  load_raw(wB, tid, rawB);
  pack_store(rawA, ldsA[0], tid);
  pack_store(rawB, ldsB[0], tid);
  __syncthreads();

  for (int kb = 0; kb < NKSTEPS; ++kb) {
    const int cur = kb & 1;
    const bool more = (kb + 1 < NKSTEPS);

    v8i af[4];
#pragma unroll
    for (int ms = 0; ms < 4; ++ms) {
      const unsigned char* ra =
          &ldsA[cur][(wm * 64 + ms * 16 + lm) * LDSTRIDE + hi * 16];
      v8i a;
#pragma unroll
      for (int j = 0; j < 4; ++j) {
        v2i d = *reinterpret_cast<const v2i*>(ra + (j & 1) * 8 + (j >> 1) * 32);
        a[2 * j] = d.x; a[2 * j + 1] = d.y;
      }
      af[ms] = a;
    }

    if (more) {
      load_raw(xA + (kb + 1) * BK, tid, rawA);
      load_raw(wB + (kb + 1) * BK, tid, rawB);
    }
    if (kb + 2 < NKSTEPS) {
      __builtin_prefetch(xA + (size_t)(tid >> 1) * KDIM + (kb + 2) * BK + (tid & 1) * 32, 0, 1);
      __builtin_prefetch(wB + (size_t)(tid >> 1) * KDIM + (kb + 2) * BK + (tid & 1) * 32, 0, 1);
    }

#pragma unroll
    for (int ns = 0; ns < 2; ++ns) {
      const unsigned char* rb =
          &ldsB[cur][(wn * 32 + ns * 16 + lm) * LDSTRIDE + hi * 16];
      v4i q0 = *reinterpret_cast<const v4i*>(rb);
      v4i q1 = *reinterpret_cast<const v4i*>(rb + 32);
      v8i b;
      b[0] = q0.x; b[1] = q0.y; b[2] = q0.z; b[3] = q0.w;
      b[4] = q1.x; b[5] = q1.y; b[6] = q1.z; b[7] = q1.w;
#pragma unroll
      for (int ms = 0; ms < 4; ++ms)
        acc[ms][ns] = __builtin_amdgcn_wmma_i32_16x16x64_iu8(
             true, af[ms],  true, b, acc[ms][ns],
             false,  false);
    }

    if (more) {
      pack_store(rawA, ldsA[cur ^ 1], tid);
      pack_store(rawB, ldsB[cur ^ 1], tid);
    }
    __syncthreads();
  }

  const float alpha = pa[0];
  const float beta  = pb[0];
#pragma unroll
  for (int ns = 0; ns < 2; ++ns) {
    const int n  = tileN + wn * 32 + ns * 16 + lm;
    const float bv = bias[n];
#pragma unroll
    for (int ms = 0; ms < 4; ++ms) {
      const int mbase = tileM + wm * 64 + ms * 16 + hi * 8;
#pragma unroll
      for (int r = 0; r < 8; ++r) {
        float y = fmaf(alpha, (float)acc[ms][ns][r], bv);
        float g = 0.5f * y * (1.0f + erff(y * 0.70710678118654752f));
        float q = __builtin_rintf(g * beta);
        q = fminf(fmaxf(q, -128.0f), 127.0f);
        out[(size_t)(mbase + r) * ODIM + n] = q;
      }
    }
  }
}

extern "C" void kernel_launch(void* const* d_in, const int* in_sizes, int n_in,
                              void* d_out, int out_size, void* d_ws, size_t ws_size,
                              hipStream_t stream) {
  (void)in_sizes; (void)n_in; (void)out_size; (void)d_ws; (void)ws_size;
  const int*   x    = (const int*)d_in[0];
  const int*   wgt  = (const int*)d_in[1];
  const float* bias = (const float*)d_in[2];
  const float* a    = (const float*)d_in[3];
  const float* b    = (const float*)d_in[4];
  float* out = (float*)d_out;

  dim3 grid(ODIM / BN, TOKENS / BM);
  dim3 block(256);
  w8a8_gelu_q_wmma<<<grid, block, 0, stream>>>(x, wgt, bias, a, b, out);
}
